// BiMambaBlock_67705864454289
// MI455X (gfx1250) — matrix operand order repaired and checked by reading, not compiled, not run on hardware
//
#include <hip/hip_runtime.h>

// ---------------------------------------------------------------------------
// BiMamba block for MI455X (gfx1250): bf16 WMMA GEMMs with double-buffered
// async-to-LDS (ASYNCcnt) staging + f32 selective scan.
// ---------------------------------------------------------------------------

#define D_MODEL 1024
#define D_STATE 16
#define D_CONV  4
#define D_INNER 2048
#define DT_RANK 64
#define BATCH   4
#define SEQ     2048
#define BL      (BATCH * SEQ)   // 8192 tokens

typedef unsigned short u16;
typedef __attribute__((ext_vector_type(16))) __bf16        v16bf;
typedef __attribute__((ext_vector_type(8)))  float         v8f;
typedef __attribute__((ext_vector_type(4)))  unsigned int  uint4v;
typedef __attribute__((ext_vector_type(4)))  float         f4v;

union FragBF { uint4v q[2]; v16bf v; };

static __device__ __forceinline__ u16 f2bf(float f) {
    unsigned int u = __float_as_uint(f);
    unsigned int r = (u + 0x7FFFu + ((u >> 16) & 1u)) >> 16;   // RNE
    return (u16)r;
}
static __device__ __forceinline__ float bf2f(u16 h) {
    return __uint_as_float(((unsigned int)h) << 16);
}
static __device__ __forceinline__ float siluf(float v) {
    return v / (1.f + __expf(-v));
}

// Async DMA: 16B per lane, global -> LDS, tracked by ASYNCcnt (no VGPR data).
// VDST = wave-relative LDS byte address; VADDR = 64-bit global address.
static __device__ __forceinline__ void async_ld_b128(const void* gsrc, void* ldst) {
    asm volatile("global_load_async_to_lds_b128 %0, %1, off"
                 :: "v"((unsigned)(size_t)ldst), "v"(gsrc)
                 : "memory");
}
// Async loads retire in order per wave: waiting ASYNCcnt<=4 releases the
// previous slab's 4 transfers while the 4 just-issued ones remain in flight.
static __device__ __forceinline__ void wait_async4(void) {
    asm volatile("s_wait_asynccnt 0x4" ::: "memory");
}
static __device__ __forceinline__ void wait_async0(void) {
    asm volatile("s_wait_asynccnt 0x0" ::: "memory");
}

// --------------------------- f32 -> bf16 convert ---------------------------
__global__ __launch_bounds__(256) void convert_bf16_kernel(
    const float* __restrict__ src, u16* __restrict__ dst, int n)
{
    int i = blockIdx.x * 256 + threadIdx.x;
    if (i < n) dst[i] = f2bf(src[i]);
}

// ------------------------------- LayerNorm ---------------------------------
__global__ __launch_bounds__(256) void layernorm_kernel(
    const float* __restrict__ x, const float* __restrict__ gamma,
    const float* __restrict__ beta, u16* __restrict__ hbf,
    u16* __restrict__ hrev)
{
    __shared__ float red[256];
    const int row = blockIdx.x;
    const int tid = threadIdx.x;
    const float* xr = x + (size_t)row * D_MODEL;
    f4v f = ((const f4v*)xr)[tid];

    float s = f[0] + f[1] + f[2] + f[3];
    red[tid] = s; __syncthreads();
    for (int o = 128; o > 0; o >>= 1) { if (tid < o) red[tid] += red[tid + o]; __syncthreads(); }
    const float mu = red[0] * (1.f / D_MODEL);
    __syncthreads();

    float d0 = f[0]-mu, d1 = f[1]-mu, d2 = f[2]-mu, d3 = f[3]-mu;
    red[tid] = d0*d0 + d1*d1 + d2*d2 + d3*d3; __syncthreads();
    for (int o = 128; o > 0; o >>= 1) { if (tid < o) red[tid] += red[tid + o]; __syncthreads(); }
    const float rstd = rsqrtf(red[0] * (1.f / D_MODEL) + 1e-5f);

    const int b = row / SEQ, t = row % SEQ;
    const size_t ro = (size_t)row * D_MODEL;
    const size_t rr = ((size_t)b * SEQ + (SEQ - 1 - t)) * D_MODEL;
    #pragma unroll
    for (int i = 0; i < 4; i++) {
        const int c = tid * 4 + i;
        const float h = (f[i] - mu) * rstd * gamma[c] + beta[c];
        const u16 hb = f2bf(h);
        hbf[ro + c] = hb;
        hrev[rr + c] = hb;
    }
}

// ------------------------- bf16 WMMA GEMM (C = A*W^T) ----------------------
// A: M x K bf16 (lda), W: N x K bf16 (ldw). 128x128x32 workgroup tile,
// 8 waves, each wave = 64x32 region = 4x2 v_wmma_f32_16x16x32_bf16 tiles.
// USE_ASYNC=1 (N % 128 == 0): double-buffered global_load_async_to_lds_b128
// pipeline — issue slab k+1, s_wait_asynccnt 4 (slab k done), barrier, MMA.
// Epilogues (compile-time):
//   0: bf16 store at [crow*ldc + colOff + n], crow optionally time-reversed
//   1: f32 store + bf16 shadow store (x_proj -> x_dbl)
//   2: softplus(v + bias[n]) -> bf16 (dt_proj -> delta)
//   3: v + bias[n] + resid[row*ldc+n] -> f32 (final output)
#define BM 128
#define BN 128
#define BK 32

template <int EPI, int USE_ASYNC>
__global__ __launch_bounds__(256) void gemm_bf16_kernel(
    const u16* __restrict__ A, int lda,
    const u16* __restrict__ W, int ldw,
    float* __restrict__ outF, u16* __restrict__ outBF, int ldc, int colOff,
    const float* __restrict__ bias, const float* __restrict__ resid,
    int N, int K, int revRows)
{
    __shared__ __align__(16) u16 As[2][BM * BK];
    __shared__ __align__(16) u16 Ws[2][BN * BK];

    const int tid  = threadIdx.x;
    const int bm   = blockIdx.y * BM;
    const int bn   = blockIdx.x * BN;
    const int wave = tid >> 5;
    const int lane = tid & 31;
    const int wm   = (wave & 1) * 64;    // wave row base inside tile
    const int wn   = (wave >> 1) * 32;   // wave col base inside tile
    const int lm   = lane & 15;
    const int kh   = lane >> 4;          // K-half selector per ISA layout

    v8f acc[4][2];
    #pragma unroll
    for (int i = 0; i < 4; i++)
        #pragma unroll
        for (int j = 0; j < 2; j++)
            #pragma unroll
            for (int q = 0; q < 8; q++) acc[i][j][q] = 0.f;

    const int ldrow  = tid >> 1;          // 0..127
    const int ldhalf = (tid & 1) * 16;    // 0 / 16 columns

    // issue one 128x32 A-tile + 128x32 W-tile slab into LDS buffer `buf`
    auto issue_slab = [&](int k0, int buf) {
        const u16* sA = A + (size_t)(bm + ldrow) * lda + k0 + ldhalf;
        const u16* sW = W + (size_t)(bn + ldrow) * ldw + k0 + ldhalf;
        u16* dA = As[buf] + ldrow * BK + ldhalf;
        u16* dW = Ws[buf] + ldrow * BK + ldhalf;
        async_ld_b128(sA,     dA);
        async_ld_b128(sA + 8, dA + 8);
        async_ld_b128(sW,     dW);
        async_ld_b128(sW + 8, dW + 8);
    };

    // consume one staged slab: 4x2 fragment WMMAs per wave
    auto mma_slab = [&](const u16* Asb, const u16* Wsb) {
        // B fragments: lane n reads Wsb[n][kh*16 .. +15] (two b128)
        FragBF bfr[2];
        #pragma unroll
        for (int j = 0; j < 2; j++) {
            const u16* p = Wsb + (wn + j * 16 + lm) * BK + kh * 16;
            bfr[j].q[0] = ((const uint4v*)p)[0];
            bfr[j].q[1] = ((const uint4v*)p)[1];
        }
        #pragma unroll
        for (int i = 0; i < 4; i++) {
            FragBF af;
            const u16* p = Asb + (wm + i * 16 + lm) * BK;
            af.q[0] = ((const uint4v*)(p + kh * 16))[0];       // K = kh*16 .. +7
            af.q[1] = ((const uint4v*)(p + kh * 16 + 8))[0];   // K = kh*16+8 .. +15
            #pragma unroll
            for (int j = 0; j < 2; j++) {
                acc[i][j] = __builtin_amdgcn_wmma_f32_16x16x32_bf16(
                    false, af.v, false, bfr[j].v,
                    (short)0, acc[i][j], false, false);
            }
        }
    };

    if (USE_ASYNC) {
        // -------- double-buffered async pipeline --------
        issue_slab(0, 0);
        for (int k0 = 0; k0 < K; k0 += BK) {
            const int cur = (k0 / BK) & 1;
            if (k0 + BK < K) {
                issue_slab(k0 + BK, cur ^ 1);   // fill other buffer
                wait_async4();                  // previous slab complete
            } else {
                wait_async0();                  // drain
            }
            __syncthreads();
            if (k0 + 2 * BK < K) {              // GL2 prefetch two slabs ahead
                __builtin_prefetch(A + (size_t)(bm + ldrow) * lda + k0 + 2 * BK, 0, 1);
                __builtin_prefetch(W + (size_t)(bn + ldrow) * ldw + k0 + 2 * BK, 0, 1);
            }
            mma_slab(As[cur], Ws[cur]);
            __syncthreads();                    // protect buffer `cur` before reuse
        }
    } else {
        // -------- register-staged path (ragged N: zero-fill past N) --------
        for (int k0 = 0; k0 < K; k0 += BK) {
            const u16* srcA = A + (size_t)(bm + ldrow) * lda + k0 + ldhalf;
            const u16* srcW = W + (size_t)(bn + ldrow) * ldw + k0 + ldhalf;
            u16* dstA = As[0] + ldrow * BK + ldhalf;
            u16* dstW = Ws[0] + ldrow * BK + ldhalf;
            {
                uint4v a0 = ((const uint4v*)srcA)[0];
                uint4v a1 = ((const uint4v*)srcA)[1];
                ((uint4v*)dstA)[0] = a0;
                ((uint4v*)dstA)[1] = a1;
            }
            {
                uint4v w0, w1;
                #pragma unroll
                for (int q = 0; q < 4; q++) { w0[q] = 0u; w1[q] = 0u; }
                if (bn + ldrow < N) {
                    w0 = ((const uint4v*)srcW)[0];
                    w1 = ((const uint4v*)srcW)[1];
                }
                ((uint4v*)dstW)[0] = w0;
                ((uint4v*)dstW)[1] = w1;
            }
            __syncthreads();
            if (k0 + BK < K) {
                __builtin_prefetch(A + (size_t)(bm + ldrow) * lda + k0 + BK, 0, 1);
                if (bn + ldrow < N)
                    __builtin_prefetch(W + (size_t)(bn + ldrow) * ldw + k0 + BK, 0, 1);
            }
            mma_slab(As[0], Ws[0]);
            __syncthreads();
        }
    }

    // ---- epilogue: C/D layout — lane<16: (M=v, N=lane); lane>=16: (M=v+8) ----
    #pragma unroll
    for (int i = 0; i < 4; i++) {
        #pragma unroll
        for (int j = 0; j < 2; j++) {
            const int ncol = bn + wn + j * 16 + lm;
            if (ncol >= N) continue;
            const int mbase = bm + wm + i * 16 + kh * 8;
            #pragma unroll
            for (int v = 0; v < 8; v++) {
                const int row = mbase + v;
                const float val = acc[i][j][v];
                if (EPI == 0) {
                    int crow = row;
                    if (revRows) {
                        const int b = row / SEQ, t = row % SEQ;
                        crow = b * SEQ + (SEQ - 1 - t);
                    }
                    outBF[(size_t)crow * ldc + colOff + ncol] = f2bf(val);
                } else if (EPI == 1) {
                    outF [(size_t)row * ldc + ncol] = val;
                    outBF[(size_t)row * ldc + ncol] = f2bf(val);
                } else if (EPI == 2) {
                    const float tt = val + bias[ncol];
                    const float sp = (tt > 20.f) ? tt : __logf(1.f + __expf(tt));
                    outBF[(size_t)row * ldc + ncol] = f2bf(sp);
                } else { // EPI == 3
                    outF[(size_t)row * ldc + ncol] =
                        val + bias[ncol] + resid[(size_t)row * ldc + ncol];
                }
            }
        }
    }
}

// ----------------------- depthwise causal conv + SiLU ----------------------
__global__ __launch_bounds__(256) void conv_silu_kernel(
    const u16* __restrict__ xz, const float* __restrict__ w,
    const float* __restrict__ bias, u16* __restrict__ xc)
{
    const int row = blockIdx.x;                       // token row (b*SEQ+t)
    const int d   = blockIdx.y * 256 + threadIdx.x;   // channel
    const int b = row / SEQ, t = row % SEQ;
    float acc = bias[d];
    #pragma unroll
    for (int j = 0; j < D_CONV; j++) {
        const int tt = t - (D_CONV - 1) + j;
        if (tt >= 0)
            acc += w[d * D_CONV + j] *
                   bf2f(xz[((size_t)b * SEQ + tt) * (2 * D_INNER) + d]);
    }
    xc[(size_t)row * D_INNER + d] = f2bf(siluf(acc));
}

// ----------------------------- selective scan ------------------------------
// One thread per (batch, channel): 16-state recurrence in registers.
// B_t/C_t (shared across channels) staged through LDS in 32-step chunks.
__global__ __launch_bounds__(256) void scan_kernel(
    const float* __restrict__ xdbl, const u16* __restrict__ delta_bf,
    const u16* __restrict__ xc_bf, const u16* __restrict__ xz_bf,
    const float* __restrict__ A_log, const float* __restrict__ Dp,
    u16* __restrict__ ybf)
{
    __shared__ float bc[32 * 32];          // 32 steps x (16 B + 16 C)
    const int b = blockIdx.x >> 3;
    const int d = (blockIdx.x & 7) * 256 + threadIdx.x;

    float Arow[D_STATE];
    #pragma unroll
    for (int s = 0; s < D_STATE; s++) Arow[s] = -__expf(A_log[d * D_STATE + s]);
    const float Dd = Dp[d];
    float h[D_STATE];
    #pragma unroll
    for (int s = 0; s < D_STATE; s++) h[s] = 0.f;

    for (int tc = 0; tc < SEQ; tc += 32) {
        #pragma unroll
        for (int i = 0; i < 4; i++) {
            const int flat = i * 256 + threadIdx.x;
            const int st = flat >> 5, col = flat & 31;
            bc[flat] = xdbl[((size_t)b * SEQ + tc + st) * 96 + DT_RANK + col];
        }
        __syncthreads();
        for (int ti = 0; ti < 32; ti++) {
            const size_t trow = (size_t)b * SEQ + tc + ti;
            const float dlt = bf2f(delta_bf[trow * D_INNER + d]);
            const float u   = bf2f(xc_bf[trow * D_INNER + d]);
            const float du  = dlt * u;
            const float* Bt = &bc[ti * 32];
            float y = 0.f;
            #pragma unroll
            for (int s = 0; s < D_STATE; s++) {
                const float e = __expf(dlt * Arow[s]);
                h[s] = e * h[s] + du * Bt[s];
                y += h[s] * Bt[16 + s];
            }
            const float zv = bf2f(xz_bf[trow * (2 * D_INNER) + D_INNER + d]);
            y = (y + u * Dd) * siluf(zv);
            ybf[trow * D_INNER + d] = f2bf(y);
        }
        __syncthreads();
    }
}

// --------------------------------- driver ----------------------------------
extern "C" void kernel_launch(void* const* d_in, const int* in_sizes, int n_in,
                              void* d_out, int out_size, void* d_ws, size_t ws_size,
                              hipStream_t stream)
{
    const float* x     = (const float*)d_in[0];
    const float* gamma = (const float*)d_in[1];
    const float* beta  = (const float*)d_in[2];
    const float* in_w[2]    = {(const float*)d_in[3],  (const float*)d_in[12]};
    const float* conv_w[2]  = {(const float*)d_in[4],  (const float*)d_in[13]};
    const float* conv_b[2]  = {(const float*)d_in[5],  (const float*)d_in[14]};
    const float* xproj_w[2] = {(const float*)d_in[6],  (const float*)d_in[15]};
    const float* dt_w[2]    = {(const float*)d_in[7],  (const float*)d_in[16]};
    const float* dt_b[2]    = {(const float*)d_in[8],  (const float*)d_in[17]};
    const float* A_log[2]   = {(const float*)d_in[9],  (const float*)d_in[18]};
    const float* Dp[2]      = {(const float*)d_in[10], (const float*)d_in[19]};
    const float* mout_w[2]  = {(const float*)d_in[11], (const float*)d_in[20]};
    const float* out_w = (const float*)d_in[21];
    const float* out_b = (const float*)d_in[22];
    float* out = (float*)d_out;

    char* ws = (char*)d_ws;
    size_t off = 0;
    auto wsalloc = [&](size_t bytes) -> char* {
        char* p = ws + off;
        off += (bytes + 255) & ~(size_t)255;
        return p;
    };

    // persistent per-call scratch
    u16* h_bf   = (u16*)wsalloc((size_t)BL * D_MODEL * 2);
    u16* h_rev  = (u16*)wsalloc((size_t)BL * D_MODEL * 2);
    u16* w_in_bf[2]    = {(u16*)wsalloc((size_t)2*D_INNER*D_MODEL*2),
                          (u16*)wsalloc((size_t)2*D_INNER*D_MODEL*2)};
    u16* w_xproj_bf[2] = {(u16*)wsalloc((size_t)96*D_INNER*2),
                          (u16*)wsalloc((size_t)96*D_INNER*2)};
    u16* w_dt_bf[2]    = {(u16*)wsalloc((size_t)D_INNER*DT_RANK*2),
                          (u16*)wsalloc((size_t)D_INNER*DT_RANK*2)};
    u16* w_mout_bf[2]  = {(u16*)wsalloc((size_t)D_MODEL*D_INNER*2),
                          (u16*)wsalloc((size_t)D_MODEL*D_INNER*2)};
    u16* w_out_bf = (u16*)wsalloc((size_t)D_MODEL*2*D_MODEL*2);
    u16* comb_bf  = (u16*)wsalloc((size_t)BL * D_INNER * 2);
    // per-direction reused scratch
    u16*   xz_bf    = (u16*)wsalloc((size_t)BL * 2 * D_INNER * 2);
    u16*   xc_bf    = (u16*)wsalloc((size_t)BL * D_INNER * 2);
    float* xdbl_f   = (float*)wsalloc((size_t)BL * 96 * 4);
    u16*   xdbl_bf  = (u16*)wsalloc((size_t)BL * 96 * 2);
    u16*   delta_bf = (u16*)wsalloc((size_t)BL * D_INNER * 2);
    u16*   y_bf     = (u16*)wsalloc((size_t)BL * D_INNER * 2);

    auto cvt = [&](const float* s, u16* d, int n) {
        convert_bf16_kernel<<<(n + 255) / 256, 256, 0, stream>>>(s, d, n);
    };
    for (int dir = 0; dir < 2; dir++) {
        cvt(in_w[dir],    w_in_bf[dir],    2 * D_INNER * D_MODEL);
        cvt(xproj_w[dir], w_xproj_bf[dir], 96 * D_INNER);
        cvt(dt_w[dir],    w_dt_bf[dir],    D_INNER * DT_RANK);
        cvt(mout_w[dir],  w_mout_bf[dir],  D_MODEL * D_INNER);
    }
    cvt(out_w, w_out_bf, D_MODEL * 2 * D_MODEL);

    layernorm_kernel<<<BL, 256, 0, stream>>>(x, gamma, beta, h_bf, h_rev);

    const dim3 blk(256);
    for (int dir = 0; dir < 2; dir++) {
        const u16* hsrc = dir ? h_rev : h_bf;

        // xz = h @ in_w^T   (M=8192, N=4096, K=1024) -> bf16
        gemm_bf16_kernel<0, 1><<<dim3(4096 / BN, BL / BM), blk, 0, stream>>>(
            hsrc, D_MODEL, w_in_bf[dir], D_MODEL,
            nullptr, xz_bf, 2 * D_INNER, 0, nullptr, nullptr,
            2 * D_INNER, D_MODEL, 0);

        // depthwise causal conv + SiLU -> xc (bf16)
        conv_silu_kernel<<<dim3(BL, D_INNER / 256), blk, 0, stream>>>(
            xz_bf, conv_w[dir], conv_b[dir], xc_bf);

        // x_dbl = xc @ xproj_w^T  (N=96, ragged -> register-staged path)
        gemm_bf16_kernel<1, 0><<<dim3(1, BL / BM), blk, 0, stream>>>(
            xc_bf, D_INNER, w_xproj_bf[dir], D_INNER,
            xdbl_f, xdbl_bf, 96, 0, nullptr, nullptr,
            96, D_INNER, 0);

        // delta = softplus(dt @ dt_w^T + dt_b)  (N=2048, K=64) -> bf16
        gemm_bf16_kernel<2, 1><<<dim3(D_INNER / BN, BL / BM), blk, 0, stream>>>(
            xdbl_bf, 96, w_dt_bf[dir], DT_RANK,
            nullptr, delta_bf, D_INNER, 0, dt_b[dir], nullptr,
            D_INNER, DT_RANK, 0);

        // selective scan (fuses +u*D and SiLU(z) gate) -> y (bf16)
        scan_kernel<<<BATCH * (D_INNER / 256), blk, 0, stream>>>(
            xdbl_f, delta_bf, xc_bf, xz_bf, A_log[dir], Dp[dir], y_bf);

        // dir-output = y @ mout_w^T -> comb columns [dir*1024, ...),
        // backward rows are written time-reversed back to natural order
        gemm_bf16_kernel<0, 1><<<dim3(D_MODEL / BN, BL / BM), blk, 0, stream>>>(
            y_bf, D_INNER, w_mout_bf[dir], D_INNER,
            nullptr, comb_bf, D_INNER, dir * D_MODEL, nullptr, nullptr,
            D_MODEL, D_INNER, dir);
    }

    // out = comb @ out_w^T + out_b + x   (M=8192, N=1024, K=2048) -> f32
    gemm_bf16_kernel<3, 1><<<dim3(D_MODEL / BN, BL / BM), blk, 0, stream>>>(
        comb_bf, D_INNER, w_out_bf, D_INNER,
        out, nullptr, D_MODEL, 0, out_b, x,
        D_MODEL, D_INNER, 0);

    (void)in_sizes; (void)n_in; (void)out_size; (void)ws_size;
}
